// CityDecoder_36593121362310
// MI455X (gfx1250) — matrix operand order repaired and checked by reading, not compiled, not run on hardware
//
#include <hip/hip_runtime.h>
#include <math.h>

#define Bb     512
#define Nn     1000
#define NPAD   1024
#define Ee     128
#define Hh     4
#define HDd    32
#define NAG    10
#define PITCH  136    // city row pitch (ushort): 272B -> conflict-free A-frag b128 loads
#define APITCH 1032   // attn row pitch (ushort): 2064B -> conflict-free A-frag b128 loads

typedef __bf16 bf16_t;
typedef bf16_t v16bf __attribute__((ext_vector_type(16)));
typedef bf16_t v8bf  __attribute__((ext_vector_type(8)));
typedef float  v8f   __attribute__((ext_vector_type(8)));
typedef float  v4f   __attribute__((ext_vector_type(4)));

struct us4 { unsigned short x, y, z, w; };

static __device__ __forceinline__ unsigned short f2bf(float f) {
  union { float f; unsigned int u; } v; v.f = f;
  unsigned int r = v.u + 0x7FFFu + ((v.u >> 16) & 1u);   // round-to-nearest-even
  return (unsigned short)(r >> 16);
}

// ---- LDS layout (total 319,776 B < 320 KB/WGP) ----
#define OFF_CITY 0
#define OFF_ATTN (NPAD * PITCH * 2)                    // 278528 : ushort[16][APITCH] bf16
#define OFF_SCORE (OFF_ATTN + 8 * APITCH * 2)          // f32[4][NPAD] ALIASES attn rows 8..15
#define OFF_VEC  (OFF_ATTN + 16 * APITCH * 2)          // 311552 : f32[4][128] B-vector source
#define OFF_QV   (OFF_VEC + 4 * Ee * 4)
#define OFF_QP   (OFF_QV + Ee * 4)
#define OFF_W    (OFF_QP + Ee * 4)                     // f32[4][128]
#define OFF_CTX  (OFF_W + 4 * Ee * 4)
#define OFF_OUT  (OFF_CTX + Ee * 4)
#define OFF_RED  (OFF_OUT + Ee * 4)                    // f32[256]
#define OFF_MX   (OFF_RED + 256 * 4)
#define OFF_RS   (OFF_MX + 4 * 4)
#define OFF_MASK (OFF_RS + 4 * 4)                      // uchar[NPAD]
#define SMEM_BYTES (OFF_MASK + NPAD)

__global__ __launch_bounds__(256)
void city_decoder_kernel(const float* __restrict__ graph_embed,
                         const float* __restrict__ city,
                         const int*   __restrict__ agent_id,
                         const int*   __restrict__ cur_pos,
                         const float* __restrict__ remain_city_ratio,
                         const float* __restrict__ remain_agent_ratio,
                         const float* __restrict__ costs,
                         const float* __restrict__ max_distance,
                         const float* __restrict__ remain_max_distance,
                         const unsigned char* __restrict__ mask,
                         const float* __restrict__ W_sc,
                         const float* __restrict__ W_dis,
                         const float* __restrict__ W_kv,
                         const float* __restrict__ W_q,
                         const float* __restrict__ W_out,
                         const float* __restrict__ W_k,
                         float* __restrict__ out)
{
  extern __shared__ unsigned char smem[];
  unsigned short* cityL   = (unsigned short*)(smem + OFF_CITY);   // [NPAD][PITCH]
  const bf16_t*   cityBf  = (const bf16_t*)(smem + OFF_CITY);
  unsigned short* attnT   = (unsigned short*)(smem + OFF_ATTN);   // [16][APITCH] bf16
  float*          scoresT = (float*)(smem + OFF_SCORE);           // [4][NPAD] (alias!)
  float*          vecL    = (float*)(smem + OFF_VEC);             // [4][128]
  float*          qL      = (float*)(smem + OFF_QV);
  float*          QL      = (float*)(smem + OFF_QP);
  float*          wL      = (float*)(smem + OFF_W);
  float*          ctxL    = (float*)(smem + OFF_CTX);
  float*          outL    = (float*)(smem + OFF_OUT);
  float*          redL    = (float*)(smem + OFF_RED);
  float*          mxL     = (float*)(smem + OFF_MX);
  float*          rsL     = (float*)(smem + OFF_RS);
  unsigned char*  maskL   = (unsigned char*)(smem + OFF_MASK);

  const int b    = blockIdx.x;
  const int tid  = threadIdx.x;
  const int lane = tid & 31;
  const int wave = tid >> 5;
  const float* cityB = city + (size_t)b * Nn * Ee;
  const float NEG_INF = -__builtin_inff();

  // ---- Phase L: stream city[b] f32 -> bf16 into LDS (the single HBM read of the big tensor)
  for (int i = tid; i < Nn * 32; i += 256) {
    int n = i >> 5, c = i & 31;
    const float4 v = ((const float4*)(cityB + (size_t)n * Ee))[c];
    __builtin_prefetch(cityB + (size_t)n * Ee + c * 4 + 2048, 0, 0);
    us4 p; p.x = f2bf(v.x); p.y = f2bf(v.y); p.z = f2bf(v.z); p.w = f2bf(v.w);
    *(us4*)&cityL[n * PITCH + c * 4] = p;
  }
  for (int i = tid; i < (NPAD - Nn) * 32; i += 256) {
    int n = Nn + (i >> 5), c = i & 31;
    us4 z = {0, 0, 0, 0};
    *(us4*)&cityL[n * PITCH + c * 4] = z;
  }
  for (int i = tid; i < NPAD; i += 256) maskL[i] = (i < Nn) ? mask[(size_t)b * Nn + i] : 1;

  // ---- Phase 0: q = step_context + graph_embed + dis  (threads 0..127)
  if (tid < Ee) {
    const int e   = tid;
    const int aid = agent_id[b];
    const float* depot = cityB + (size_t)(aid + 1) * Ee;
    const float* cur   = cityB + (size_t)cur_pos[b] * Ee;
    float acc = graph_embed[(size_t)b * Ee + e];
    for (int j = 0; j < Ee; ++j) {
      acc += depot[j] * W_sc[j * Ee + e];
      acc += cur[j]   * W_sc[(Ee + j) * Ee + e];
    }
    acc += remain_agent_ratio[b] * W_sc[256 * Ee + e];
    acc += remain_city_ratio[b]  * W_sc[257 * Ee + e];
    acc += costs[b * NAG + aid]   * W_dis[0 * Ee + e];
    acc += max_distance[b]        * W_dis[1 * Ee + e];
    acc += remain_max_distance[b] * W_dis[2 * Ee + e];
    qL[e] = acc;
  }
  __syncthreads();

  // ---- Q = q @ W_q
  if (tid < Ee) {
    float acc = 0.f;
    for (int j = 0; j < Ee; ++j) acc += qL[j] * W_q[j * Ee + tid];
    QL[tid] = acc;
  }
  __syncthreads();

  // ---- qk[h] = W_kv[:, hHD:(h+1)HD] @ Q_h   (K-projection folded into the query)
  if (tid < Ee) {
    const int e = tid;
    for (int h = 0; h < Hh; ++h) {
      float acc = 0.f;
      const float* wrow = W_kv + e * (2 * Ee) + h * HDd;
      const float* qh   = QL + h * HDd;
      for (int d = 0; d < HDd; ++d) acc += wrow[d] * qh[d];
      vecL[h * Ee + e] = acc;
    }
  }
  __syncthreads();

  // ---- WMMA lane geometry (wave32, 16x16x32 bf16; contiguous k order used here, same for A and B)
  const int acol  = lane & 15;
  const int abase = (lane < 16) ? 0 : 16;   // A: K in 0..15 / 16..31
  const int bbase = (lane < 16) ? 0 : 16;   // B: K in 0..15 / 16..31, col = lane&15
  const int rb    = (lane < 16) ? 0 : 8;    // C/D: rows 0..7 / 8..15

  v16bf bf[4];
  #pragma unroll
  for (int kk = 0; kk < 4; ++kk) {
    const float* vc = vecL + acol * Ee + kk * 32 + bbase;
    #pragma unroll
    for (int j = 0; j < 16; ++j)
      bf[kk][j] = (acol < Hh) ? (bf16_t)vc[j] : (bf16_t)0.f;
  }

  // ---- Pass 1: scores = city @ qk^T   (WMMA; 64 tiles strided over 8 waves)
  const float RSQ_HD = 0.17677669529663687f;  // 1/sqrt(32)
  for (int t = wave; t < NPAD / 16; t += 8) {
    const int row0  = t * 16;
    const int rbase = (row0 + acol) * PITCH + abase;
    v16bf a[4];
    #pragma unroll
    for (int kk = 0; kk < 4; ++kk) {           // issue all 8 b128 loads up front
      v8bf lo = *(const v8bf*)&cityL[rbase + kk * 32];
      v8bf hi = *(const v8bf*)&cityL[rbase + kk * 32 + 8];
      a[kk] = __builtin_shufflevector(lo, hi, 0,1,2,3,4,5,6,7,8,9,10,11,12,13,14,15);
    }
    v8f c = {};
    #pragma unroll
    for (int kk = 0; kk < 4; ++kk)
      c = __builtin_amdgcn_wmma_f32_16x16x32_bf16(false, a[kk], false, bf[kk],
                                                  (short)0, c, false, false);
    if (acol < Hh) {
      const int n0 = row0 + rb;
      const unsigned long long mq = *(const unsigned long long*)&maskL[n0];
      v4f s0, s1;
      #pragma unroll
      for (int r = 0; r < 4; ++r)
        s0[r] = ((mq >> (8 * r)) & 0xff) ? NEG_INF : c[r] * RSQ_HD;
      #pragma unroll
      for (int r = 0; r < 4; ++r)
        s1[r] = ((mq >> (8 * (r + 4))) & 0xff) ? NEG_INF : c[r + 4] * RSQ_HD;
      float* sp = &scoresT[acol * NPAD + n0];
      *(v4f*)sp = s0;                           // two ds_store_b128, contiguous
      *(v4f*)(sp + 4) = s1;
    }
  }
  __syncthreads();

  // ---- Phase 2: softmax over n per head; exp -> attnT rows 0..3 as bf16
  {
    const int h = tid & 3;
    float m = NEG_INF;
    for (int n = (tid >> 2); n < NPAD; n += 64) m = fmaxf(m, scoresT[h * NPAD + n]);
    redL[tid] = m;
    __syncthreads();
    if (tid < 4) {
      float mm = NEG_INF;
      for (int q2 = 0; q2 < 64; ++q2) mm = fmaxf(mm, redL[q2 * 4 + tid]);
      mxL[tid] = mm;
    }
    __syncthreads();
    const float mx = mxL[h];
    float s = 0.f;
    for (int n = (tid >> 2); n < NPAD; n += 64) {
      float ev = __expf(scoresT[h * NPAD + n] - mx);
      attnT[h * APITCH + n] = f2bf(ev);         // un-normalized attn (1/sum folded later)
      s += ev;
    }
    redL[tid] = s;
    __syncthreads();
    if (tid < 4) {
      float ss = 0.f;
      for (int q2 = 0; q2 < 64; ++q2) ss += redL[q2 * 4 + tid];
      rsL[tid] = 1.0f / ss;
    }
    __syncthreads();
  }

  // ---- zero attn rows 4..15 (reclaims the scoresT alias region as WMMA zero rows)
  {
    unsigned int* az = (unsigned int*)(attnT + 4 * APITCH);
    for (int i = tid; i < 6 * APITCH; i += 256) az[i] = 0u;  // 12 rows * APITCH ushorts
  }
  __syncthreads();

  // ---- Phase 3: w = attn^T @ city  (WMMA; wave w owns e-columns [16w,16w+16))
  {
    const int ecol = wave * 16 + acol;
    v8f d = {};
    #pragma unroll 2
    for (int ch = 0; ch < NPAD / 32; ++ch) {
      const int ro = acol * APITCH + ch * 32 + abase;
      v8bf lo = *(const v8bf*)&attnT[ro];
      v8bf hi = *(const v8bf*)&attnT[ro + 8];
      v16bf a = __builtin_shufflevector(lo, hi, 0,1,2,3,4,5,6,7,8,9,10,11,12,13,14,15);
      const int kb = ch * 32 + bbase;
      v16bf bb;
      #pragma unroll
      for (int j = 0; j < 16; ++j)
        bb[j] = cityBf[(kb + j) * PITCH + ecol];  // lanes span consecutive e: conflict-free
      d = __builtin_amdgcn_wmma_f32_16x16x32_bf16(false, a, false, bb,
                                                  (short)0, d, false, false);
    }
    if (lane < 16) {
      #pragma unroll
      for (int r = 0; r < Hh; ++r)
        wL[r * Ee + ecol] = d[r] * rsL[r];        // normalize here
    }
  }
  __syncthreads();

  // ---- Phase 4: ctx = w @ W_v ; out = ctx @ W_out ; u = W_k @ out
  if (tid < Ee) {
    const int h = tid >> 5, dd = tid & 31;
    float acc = 0.f;
    const float* wr = wL + h * Ee;
    for (int e2 = 0; e2 < Ee; ++e2) acc += wr[e2] * W_kv[e2 * (2 * Ee) + Ee + h * HDd + dd];
    ctxL[tid] = acc;
  }
  __syncthreads();
  if (tid < Ee) {
    float acc = 0.f;
    for (int j = 0; j < Ee; ++j) acc += ctxL[j] * W_out[j * Ee + tid];
    outL[tid] = acc;
  }
  __syncthreads();
  if (tid < Ee) {
    float acc = 0.f;
    const float* wkr = W_k + tid * Ee;
    for (int j = 0; j < Ee; ++j) acc += outL[j] * wkr[j];
    vecL[tid] = acc;                              // u -> B-source row 0
  }
  __syncthreads();

  // ---- Pass 5: logits = tanh(city @ u / sqrt(E)) * 10, masked (WMMA)
  v16bf uf[4];
  #pragma unroll
  for (int kk = 0; kk < 4; ++kk) {
    const float* vc = vecL + kk * 32 + bbase;
    #pragma unroll
    for (int j = 0; j < 16; ++j)
      uf[kk][j] = (acol == 0) ? (bf16_t)vc[j] : (bf16_t)0.f;
  }
  const float RSQ_E = 0.08838834764831845f;  // 1/sqrt(128)
  for (int t = wave; t < NPAD / 16; t += 8) {
    const int row0  = t * 16;
    const int rbase = (row0 + acol) * PITCH + abase;
    v16bf a[4];
    #pragma unroll
    for (int kk = 0; kk < 4; ++kk) {
      v8bf lo = *(const v8bf*)&cityL[rbase + kk * 32];
      v8bf hi = *(const v8bf*)&cityL[rbase + kk * 32 + 8];
      a[kk] = __builtin_shufflevector(lo, hi, 0,1,2,3,4,5,6,7,8,9,10,11,12,13,14,15);
    }
    v8f c = {};
    #pragma unroll
    for (int kk = 0; kk < 4; ++kk)
      c = __builtin_amdgcn_wmma_f32_16x16x32_bf16(false, a[kk], false, uf[kk],
                                                  (short)0, c, false, false);
    if (acol == 0) {
      const int n0 = row0 + rb;
      if (n0 + 7 < Nn) {
        const unsigned long long mq = *(const unsigned long long*)&maskL[n0];
        v4f o0, o1;
        #pragma unroll
        for (int r = 0; r < 4; ++r)
          o0[r] = ((mq >> (8 * r)) & 0xff) ? NEG_INF : tanhf(c[r] * RSQ_E) * 10.0f;
        #pragma unroll
        for (int r = 0; r < 4; ++r)
          o1[r] = ((mq >> (8 * (r + 4))) & 0xff) ? NEG_INF : tanhf(c[r + 4] * RSQ_E) * 10.0f;
        float* op = out + (size_t)b * Nn + n0;
        *(v4f*)op = o0;                         // global_store_b128 x2
        *(v4f*)(op + 4) = o1;
      } else {
        #pragma unroll
        for (int r = 0; r < 8; ++r) {
          const int n = n0 + r;
          if (n < Nn) {
            float lg = tanhf(c[r] * RSQ_E) * 10.0f;
            out[(size_t)b * Nn + n] = maskL[n] ? NEG_INF : lg;
          }
        }
      }
    }
  }
}

extern "C" void kernel_launch(void* const* d_in, const int* in_sizes, int n_in,
                              void* d_out, int out_size, void* d_ws, size_t ws_size,
                              hipStream_t stream) {
  (void)in_sizes; (void)n_in; (void)d_ws; (void)ws_size; (void)out_size;
  const float* graph_embed         = (const float*)d_in[0];
  const float* city_embed          = (const float*)d_in[1];
  const int*   agent_id            = (const int*)d_in[2];
  const int*   cur_pos             = (const int*)d_in[3];
  const float* remain_city_ratio   = (const float*)d_in[4];
  const float* remain_agent_ratio  = (const float*)d_in[5];
  const float* costs               = (const float*)d_in[6];
  const float* max_distance        = (const float*)d_in[7];
  const float* remain_max_distance = (const float*)d_in[8];
  const unsigned char* mask        = (const unsigned char*)d_in[9];
  const float* W_sc  = (const float*)d_in[10];
  const float* W_dis = (const float*)d_in[11];
  const float* W_kv  = (const float*)d_in[12];
  const float* W_q   = (const float*)d_in[13];
  const float* W_out = (const float*)d_in[14];
  const float* W_k   = (const float*)d_in[15];

  (void)hipFuncSetAttribute((const void*)city_decoder_kernel,
                            hipFuncAttributeMaxDynamicSharedMemorySize,
                            (int)SMEM_BYTES);
  city_decoder_kernel<<<Bb, 256, SMEM_BYTES, stream>>>(
      graph_embed, city_embed, agent_id, cur_pos, remain_city_ratio,
      remain_agent_ratio, costs, max_distance, remain_max_distance, mask,
      W_sc, W_dis, W_kv, W_q, W_out, W_k, (float*)d_out);
}
